// DecorrelateLossClass_2147483648069
// MI455X (gfx1250) — matrix operand order repaired and checked by reading, not compiled, not run on hardware
//
#include <hip/hip_runtime.h>
#include <hip/hip_bf16.h>

#define NSAMP 16384
#define DIM   512
#define NCLS  128
#define EPSV  1e-8f
#define PSTRIDE 20480   // >= 16384 + 128*31 padded rows, multiple of 32

typedef __attribute__((ext_vector_type(16))) _Float16 v16h;
typedef __attribute__((ext_vector_type(8)))  _Float16 v8h;
typedef __attribute__((ext_vector_type(8)))  float    v8f;

union ABfrag { v16h v; v8h h[2]; };

// ---- workspace layout (bytes) ----
#define OFF_COUNTS 0u                    // 128 * 4
#define OFF_POS    512u                  // 16384 * 4
#define OFF_POFF   66048u                // 129 * 4 (padded to 1024)
#define OFF_IDX    67072u                // 16384 * 4
#define OFF_MEAN   132608u               // 128*512*4
#define OFF_INVSTD 394752u               // 128*512*4
#define OFF_ACCUM  656896u               // 128 * 4
#define OFF_XHT    1048576u              // 512 * PSTRIDE * 2  (~20 MB)
#define XHT_BYTES  ((size_t)DIM * PSTRIDE * 2u)

// 1) Deterministic per-class rank + counts. One wave per class.
__global__ __launch_bounds__(32) void rank_kernel(const int* __restrict__ y,
                                                  int* __restrict__ counts,
                                                  int* __restrict__ pos) {
    const int k = blockIdx.x;
    const int lane = threadIdx.x;
    int cnt = 0;
    for (int base = 0; base < NSAMP; base += 32) {
        const int i = base + lane;
        const bool m = (y[i] == k);
        unsigned long long mask = __ballot(m);
        const unsigned long long below = mask & ((1ull << lane) - 1ull);
        if (m) pos[i] = cnt + (int)__popcll(below);
        cnt += (int)__popcll(mask);
    }
    if (lane == 0) counts[k] = cnt;
}

// 2) Padded (multiple-of-32) class offsets. Trivial serial scan over 128.
__global__ void poff_kernel(const int* __restrict__ counts, int* __restrict__ poff) {
    if (threadIdx.x == 0 && blockIdx.x == 0) {
        int acc = 0;
        poff[0] = 0;
        for (int k = 0; k < NCLS; ++k) {
            acc += (counts[k] + 31) & ~31;
            poff[k + 1] = acc;
        }
    }
}

// 3) Scatter row ids to class-grouped slots.
__global__ void scatter_kernel(const int* __restrict__ y, const int* __restrict__ pos,
                               const int* __restrict__ poff, int* __restrict__ idxrow) {
    const int i = blockIdx.x * blockDim.x + threadIdx.x;
    if (i < NSAMP) idxrow[poff[y[i]] + pos[i]] = i;
}

// 4) Per-class mean / inverse-std. Block = class, thread = feature dim.
__global__ __launch_bounds__(512) void stats_kernel(const float* __restrict__ x,
                                                    const int* __restrict__ counts,
                                                    const int* __restrict__ poff,
                                                    const int* __restrict__ idxrow,
                                                    float* __restrict__ meanA,
                                                    float* __restrict__ invstdA) {
    const int k = blockIdx.x;
    const int d = threadIdx.x;
    const int n = counts[k];
    const int base = poff[k];
    float s1 = 0.f, s2 = 0.f;
    for (int r = 0; r < n; ++r) {
        const int row = idxrow[base + r];
        const float v = x[(size_t)row * DIM + d];
        s1 += v;
        s2 += v * v;
    }
    const float fn = (float)n;
    const float mean = s1 / fmaxf(fn, 1.f);
    float var = (s2 - fn * mean * mean) / fmaxf(fn - 1.f, 1.f);
    var = fmaxf(var, 0.f);
    meanA[(size_t)k * DIM + d] = mean;
    invstdA[(size_t)k * DIM + d] = 1.f / sqrtf(EPSV + var);
}

// 5) Normalize and write transposed f16 matrix: xhT[dim][padded_slot].
__global__ __launch_bounds__(256) void norm_kernel(const float* __restrict__ x,
                                                   const int* __restrict__ y,
                                                   const int* __restrict__ pos,
                                                   const int* __restrict__ poff,
                                                   const float* __restrict__ meanA,
                                                   const float* __restrict__ invstdA,
                                                   _Float16* __restrict__ xhT) {
    const int i = blockIdx.x;
    const int k = y[i];
    const int slot = poff[k] + pos[i];
    for (int d = threadIdx.x; d < DIM; d += blockDim.x) {
        const float v = (x[(size_t)i * DIM + d] - meanA[(size_t)k * DIM + d]) *
                        invstdA[(size_t)k * DIM + d];
        xhT[(size_t)d * PSTRIDE + slot] = (_Float16)v;
    }
}

// 6) WMMA Gram: one wave per 32x32 super-tile per class, upper triangle only.
//    Register-blocked 2x2 subtiles: 4 v_wmma per K-step, fragments reused.
//    G = Xn_k^T Xn_k ; accumulate ||G||_F^2 - ||diag(G)||^2 per class, using
//    symmetry: full = diag supertiles + 2 * upper off-diag supertiles.
__global__ __launch_bounds__(32) void gram_kernel(const _Float16* __restrict__ xhT,
                                                  const int* __restrict__ counts,
                                                  const int* __restrict__ poff,
                                                  float* __restrict__ accum) {
    const int bx = blockIdx.x;            // M super-tile (32 dims)
    const int by = blockIdx.y;            // N super-tile (32 dims)
    if (bx > by) return;                  // symmetry: upper triangle only (uniform exit)
    const int k = blockIdx.z;
    const int nk = counts[k];
    if (nk <= 1) return;                  // invalid class (counts <= REJECT_THRESHOLD)
    const int base = poff[k];
    const int nkpad = poff[k + 1] - base; // multiple of 32
    const int lane = threadIdx.x;
    const bool diagSuper = (bx == by);
    const float w = diagSuper ? 1.f : 2.f;

    // A fragment (16x32 f16): lanes 0-15 hold K 0..7 / 16..23; lanes 16-31 K 8..15 / 24..31
    const int aSel = (lane >> 4) ? 8 : 0;
    // B fragment (32x16 f16): lanes 0-15 hold K 0..7 / 16..23; lanes 16-31 K 8..15 / 24..31
    const int bSel = (lane >> 4) ? 8 : 0;
    const int mA = bx * 32 + (lane & 15);
    const int nB = by * 32 + (lane & 15);
    const _Float16* A0 = xhT + (size_t)mA * PSTRIDE + base + aSel;           // M subtile 0
    const _Float16* A1 = A0 + (size_t)16 * PSTRIDE;                          // M subtile 1
    const _Float16* B0 = xhT + (size_t)nB * PSTRIDE + base + bSel;           // N subtile 0
    const _Float16* B1 = B0 + (size_t)16 * PSTRIDE;                          // N subtile 1

    v8f c00 = {}, c01 = {}, c10 = {}, c11 = {};
    for (int kk = 0; kk < nkpad; kk += 32) {
        ABfrag a0, a1, b0, b1;
        a0.h[0] = *(const v8h*)(A0 + kk);
        a0.h[1] = *(const v8h*)(A0 + kk + 16);
        a1.h[0] = *(const v8h*)(A1 + kk);
        a1.h[1] = *(const v8h*)(A1 + kk + 16);
        b0.h[0] = *(const v8h*)(B0 + kk);
        b0.h[1] = *(const v8h*)(B0 + kk + 16);
        b1.h[0] = *(const v8h*)(B1 + kk);
        b1.h[1] = *(const v8h*)(B1 + kk + 16);
        c00 = __builtin_amdgcn_wmma_f32_16x16x32_f16(false, a0.v, false, b0.v, (short)0, c00, false, false);
        c01 = __builtin_amdgcn_wmma_f32_16x16x32_f16(false, a0.v, false, b1.v, (short)0, c01, false, false);
        c10 = __builtin_amdgcn_wmma_f32_16x16x32_f16(false, a1.v, false, b0.v, (short)0, c10, false, false);
        c11 = __builtin_amdgcn_wmma_f32_16x16x32_f16(false, a1.v, false, b1.v, (short)0, c11, false, false);
    }

    // C layout: VGPR j -> M = j (+8 for lanes 16-31), N = lane & 15.
    // Global diagonal lives only in diag supertiles, subtiles (0,0) and (1,1).
    const int mrow = (lane >> 4) * 8;
    const int ncol = lane & 15;
    float ss = 0.f, dd = 0.f;
#pragma unroll
    for (int j = 0; j < 8; ++j) {
        const float v00 = c00[j], v01 = c01[j], v10 = c10[j], v11 = c11[j];
        ss += v00 * v00 + v01 * v01 + v10 * v10 + v11 * v11;
        if (diagSuper && (j + mrow) == ncol) dd += v00 * v00 + v11 * v11;
    }
    float off = w * ss - dd;              // dd nonzero only when w == 1
#pragma unroll
    for (int s = 16; s > 0; s >>= 1) off += __shfl_down(off, s, 32);
    if (lane == 0) atomicAdd(&accum[k], off);
}

// 7) Final scalar combine. NOTE: off_denom uses C = feature dim (512), per reference.
__global__ void finalize_kernel(const int* __restrict__ counts,
                                const float* __restrict__ accum,
                                float* __restrict__ out) {
    if (threadIdx.x == 0 && blockIdx.x == 0) {
        float ncnt = 0.f, loss_sum = 0.f;
        for (int k = 0; k < NCLS; ++k) {
            const int c = counts[k];
            if (c > 1) {  // REJECT_THRESHOLD = 1
                ncnt += (float)c;
                loss_sum += accum[k];
            }
        }
        const float off_denom = (float)DIM * (float)(DIM - 1);  // 512*511
        const float loss = loss_sum / off_denom;
        out[0] = (ncnt > 0.f) ? loss / fmaxf(ncnt, 1.f) : 0.f;
    }
}

extern "C" void kernel_launch(void* const* d_in, const int* in_sizes, int n_in,
                              void* d_out, int out_size, void* d_ws, size_t ws_size,
                              hipStream_t stream) {
    const float* x = (const float*)d_in[0];
    const int*   y = (const int*)d_in[1];
    float* out = (float*)d_out;

    char* ws = (char*)d_ws;
    int*       counts  = (int*)(ws + OFF_COUNTS);
    int*       pos     = (int*)(ws + OFF_POS);
    int*       poff    = (int*)(ws + OFF_POFF);
    int*       idxrow  = (int*)(ws + OFF_IDX);
    float*     meanA   = (float*)(ws + OFF_MEAN);
    float*     invstdA = (float*)(ws + OFF_INVSTD);
    float*     accum   = (float*)(ws + OFF_ACCUM);
    _Float16*  xhT     = (_Float16*)(ws + OFF_XHT);

    // Zero padded f16 matrix and per-class accumulators every call
    // (harness poisons d_ws; padding rows must be zero for WMMA K-loop).
    hipMemsetAsync(xhT, 0, XHT_BYTES, stream);
    hipMemsetAsync(accum, 0, NCLS * sizeof(float), stream);

    rank_kernel<<<NCLS, 32, 0, stream>>>(y, counts, pos);
    poff_kernel<<<1, 1, 0, stream>>>(counts, poff);
    scatter_kernel<<<NSAMP / 256, 256, 0, stream>>>(y, pos, poff, idxrow);
    stats_kernel<<<NCLS, DIM, 0, stream>>>(x, counts, poff, idxrow, meanA, invstdA);
    norm_kernel<<<NSAMP, 256, 0, stream>>>(x, y, pos, poff, meanA, invstdA, xhT);

    dim3 gridG(DIM / 32, DIM / 32, NCLS);  // 16 x 16 super-tiles x 128 classes
    gram_kernel<<<gridG, 32, 0, stream>>>(xhT, counts, poff, accum);

    finalize_kernel<<<1, 1, 0, stream>>>(counts, accum, out);
}
